// HTN_79894981640653
// MI455X (gfx1250) — matrix operand order repaired and checked by reading, not compiled, not run on hardware
//
#include <hip/hip_runtime.h>
#include <hip/hip_bf16.h>

typedef __attribute__((ext_vector_type(16))) _Float16 v16h;
typedef __attribute__((ext_vector_type(8)))  float    v8f;
typedef __attribute__((ext_vector_type(2)))  _Float16 h2;
typedef __attribute__((ext_vector_type(4)))  unsigned int u32x4;
typedef __attribute__((ext_vector_type(8)))  int      i32x8;
typedef __attribute__((ext_vector_type(4)))  int      i32x4;

#if defined(__has_builtin)
#  if __has_builtin(__builtin_amdgcn_tensor_load_to_lds)
#    define HAVE_TDM 1
#  endif
#endif

constexpr int Nn   = 50000;
constexpr int Ee   = 400000;
constexpr int IN0  = 128;
constexpr int Hh   = 4;
constexpr int Ff   = 64;
constexpr int HF   = 256;   // H*F
constexpr int K3F  = 192;   // 3*F
constexpr int MLPH = 64;

// ---------------- elementwise converters ----------------

__global__ __launch_bounds__(256) void k_f32_to_f16(const float* __restrict__ in,
                                                    _Float16* __restrict__ out, int n) {
  int i = blockIdx.x * 256 + threadIdx.x;
  if (i < n) out[i] = (_Float16)in[i];
}

// transpose [K, C] f32 -> [C, K] f16   (wT[n*K + k] = w[k*C + n])
__global__ __launch_bounds__(256) void k_transpose_cvt(const float* __restrict__ w,
                                                       _Float16* __restrict__ wT,
                                                       int K, int C) {
  int i = blockIdx.x * 256 + threadIdx.x;
  if (i >= K * C) return;
  int n = i / K;
  int k = i - n * K;
  wT[i] = (_Float16)w[k * C + n];
}

// ---------------- projection GEMM: C[rows,256] = A[rows,K] @ B[K,256] ----------------
// One wave per 16-row tile; 2 passes x 8 accumulators (A fragments loaded once/pass).
// Writes both f32 (for scatter/finalize) and f16 (for the edge-kernel gathers).
__global__ __launch_bounds__(256) void k_proj_gemm(const _Float16* __restrict__ A,
                                                   const _Float16* __restrict__ BT, // [256][K]
                                                   float* __restrict__ Cm,          // [rows][256]
                                                   _Float16* __restrict__ Cm16,     // [rows][256]
                                                   int rows, int K) {
  const int lane = threadIdx.x & 31;
  const int wave = threadIdx.x >> 5;
  const int hfl  = lane >> 4;
  const int ml   = lane & 15;
  const int numTiles = rows >> 4;
  const int tile = blockIdx.x * 8 + wave;
  if (tile >= numTiles) return;
  const int rowbase = tile * 16;
  const int row = rowbase + ml;
  const int ksteps = K >> 5;

  for (int pass = 0; pass < 2; ++pass) {
    v8f acc[8] = {};
    for (int kk = 0; kk < ksteps; ++kk) {
      v16h a;
#pragma unroll
      for (int v = 0; v < 8; ++v) {
        const int kA = (kk << 5) + 2 * v + ((v & 4) << 1) + (hfl << 3);
        h2 p = *(const h2*)(A + (size_t)row * K + kA);
        a[2 * v] = p[0]; a[2 * v + 1] = p[1];
      }
#pragma unroll
      for (int t = 0; t < 8; ++t) {
        v16h b;
        const int n = pass * 128 + t * 16 + ml;
#pragma unroll
        for (int v = 0; v < 8; ++v) {
          const int kB = (kk << 5) + 2 * v + ((v & 4) << 1) + (hfl << 3);
          h2 p = *(const h2*)(BT + (size_t)n * K + kB);
          b[2 * v] = p[0]; b[2 * v + 1] = p[1];
        }
        acc[t] = __builtin_amdgcn_wmma_f32_16x16x32_f16(
            false, a, false, b, (short)0, acc[t], false, false);
      }
    }
#pragma unroll
    for (int t = 0; t < 8; ++t)
#pragma unroll
      for (int r = 0; r < 8; ++r) {
        const size_t idx = (size_t)(rowbase + r + hfl * 8) * HF + pass * 128 + t * 16 + ml;
        Cm[idx]   = acc[t][r];
        Cm16[idx] = (_Float16)acc[t][r];
      }
  }
}

// ---------------- edge attention kernel ----------------
// Rows = edge-head pairs (row = e*4 + h). One wave computes 16 rows per tile:
// hidden[16,64] = comb[16,192] @ w1, fused relu, @w2, +b2, leaky-relu,
// per-edge softmax over the 4 heads, writes att[E*H]. Grid-stride over tiles.
__global__ __launch_bounds__(256) void k_edge_att(const _Float16* __restrict__ proj16,
                                                  const int* __restrict__ ei,
                                                  const _Float16* __restrict__ w1T, // [64][192]
                                                  const float* __restrict__ b1,
                                                  const float* __restrict__ w2,
                                                  const float* __restrict__ b2p,
                                                  float* __restrict__ att,
                                                  int numTiles) {
  __shared__ _Float16 w1s[MLPH * K3F];     // 24 KB; only LDS alloc -> offset 0

#ifdef HAVE_TDM
  if (threadIdx.x < 32) {                  // one TDM issue per workgroup (wave 0)
    const unsigned long long ga = (unsigned long long)(const void*)w1T;
    u32x4 g0;
    g0[0] = 1u;                                                // count=1 (valid), user mode
    g0[1] = 0u;                                                // lds_addr = 0
    g0[2] = (unsigned)ga;                                      // global_addr[31:0]
    g0[3] = (unsigned)((ga >> 32) & 0x01FFFFFFu) | (2u << 30); // global_addr[56:32] | type=2
    i32x8 g1;
    g1[0] = (1 << 16);      // workgroup_mask=0 | data_size=1 (2 bytes)
    g1[1] = (K3F << 16);    // tensor_dim0 = 192 (low 16 bits at [63:48])
    g1[2] = (MLPH << 16);   // tensor_dim0 hi=0 | tensor_dim1 = 64
    g1[3] = (K3F << 16);    // tensor_dim1 hi=0 | tile_dim0 = 192
    g1[4] = MLPH;           // tile_dim1 = 64 | tile_dim2 = 0
    g1[5] = K3F;            // tensor_dim0_stride = 192 (lo 32)
    g1[6] = 0;              // stride hi | tensor_dim1_stride lo
    g1[7] = 0;
    i32x4 gz = {0, 0, 0, 0};
#if __clang_major__ >= 23
    i32x8 gz8 = {0, 0, 0, 0, 0, 0, 0, 0};
    __builtin_amdgcn_tensor_load_to_lds(g0, g1, gz, gz, gz8, 0);
#else
    __builtin_amdgcn_tensor_load_to_lds(g0, g1, gz, gz, 0);
#endif
    __builtin_amdgcn_s_wait_tensorcnt(0);
  }
#else
  for (int i = threadIdx.x; i < (MLPH * K3F) / 2; i += 256)
    ((unsigned int*)w1s)[i] = ((const unsigned int*)w1T)[i];
#endif
  __syncthreads();

  const int lane = threadIdx.x & 31;
  const int wave = threadIdx.x >> 5;
  const int hfl  = lane >> 4;
  const int ml   = lane & 15;
  const int tileStride = gridDim.x * 8;
  const float b2 = b2p[0];

  for (int tile = blockIdx.x * 8 + wave; tile < numTiles; tile += tileStride) {
    const int rowbase = tile * 16;
    const int row = rowbase + ml;          // lane's A-matrix row
    const int e = row >> 2;
    const int h = row & 3;
    const int n0 = ei[e];
    const int n1 = ei[Ee + e];
    const int n2 = ei[2 * Ee + e];

    v8f acc[4] = {};
#pragma unroll
    for (int kk = 0; kk < 6; ++kk) {
      const int seg = kk >> 1;
      const int fb  = (kk & 1) * 32;
      const int node = (seg == 0) ? n0 : (seg == 1) ? n1 : n2;
      const _Float16* p = proj16 + (size_t)node * HF + h * Ff + fb;
      v16h a;
#pragma unroll
      for (int v = 0; v < 8; ++v) {
        const int kA = 2 * v + ((v & 4) << 1) + (hfl << 3);
        h2 pa = *(const h2*)(p + kA);
        a[2 * v] = pa[0]; a[2 * v + 1] = pa[1];
      }
#pragma unroll
      for (int t = 0; t < 4; ++t) {
        v16h b;
        const int n = t * 16 + ml;
#pragma unroll
        for (int v = 0; v < 8; ++v) {
          const int kB = kk * 32 + 2 * v + ((v & 4) << 1) + (hfl << 3);
          h2 pb = *(const h2*)(w1s + n * K3F + kB);
          b[2 * v] = pb[0]; b[2 * v + 1] = pb[1];
        }
        acc[t] = __builtin_amdgcn_wmma_f32_16x16x32_f16(
            false, a, false, b, (short)0, acc[t], false, false);
      }
    }

    // fused relu(hidden + b1) @ w2 : per-lane partial over its n column
    float partial[8];
#pragma unroll
    for (int r = 0; r < 8; ++r) partial[r] = 0.f;
#pragma unroll
    for (int t = 0; t < 4; ++t) {
      const float b1v = b1[t * 16 + ml];
      const float w2v = w2[t * 16 + ml];
#pragma unroll
      for (int r = 0; r < 8; ++r) {
        const float hv = acc[t][r] + b1v;
        partial[r] += fmaxf(hv, 0.f) * w2v;
      }
    }
    // reduce across the 16 lanes of each half-wave
#pragma unroll
    for (int off = 1; off < 16; off <<= 1)
#pragma unroll
      for (int r = 0; r < 8; ++r)
        partial[r] += __shfl_xor(partial[r], off, 32);

    float s[8];
#pragma unroll
    for (int r = 0; r < 8; ++r) {
      const float v = partial[r] + b2;
      s[r] = v > 0.f ? v : 0.2f * v;       // leaky_relu(0.2)
    }
    // softmax over heads: rows {0..3} = one edge, {4..7} = next edge
    const float m0 = fmaxf(fmaxf(s[0], s[1]), fmaxf(s[2], s[3]));
    const float m1 = fmaxf(fmaxf(s[4], s[5]), fmaxf(s[6], s[7]));
    float av[8];
#pragma unroll
    for (int r = 0; r < 4; ++r) av[r] = expf(s[r] - m0);
#pragma unroll
    for (int r = 4; r < 8; ++r) av[r] = expf(s[r] - m1);
    const float inv0 = 1.f / (av[0] + av[1] + av[2] + av[3]);
    const float inv1 = 1.f / (av[4] + av[5] + av[6] + av[7]);
#pragma unroll
    for (int r = 0; r < 4; ++r) av[r] *= inv0;
#pragma unroll
    for (int r = 4; r < 8; ++r) av[r] *= inv1;
#pragma unroll
    for (int r = 0; r < 8; ++r)
      if (ml == r) att[rowbase + hfl * 8 + r] = av[r];
  }
}

// ---------------- scatter: wsum[ei0] += att * (cj * ck) ----------------
__global__ __launch_bounds__(256) void k_scatter(const int* __restrict__ ei,
                                                 const float* __restrict__ att,
                                                 const float* __restrict__ proj,
                                                 float* __restrict__ wsum) {
  const int e = blockIdx.x;
  const int t = threadIdx.x;
  const int h = t >> 6, f = t & 63;
  const int ni = ei[e];
  const int nj = ei[Ee + e];
  const int nk = ei[2 * Ee + e];
  const float a  = att[e * Hh + h];
  const float cj = proj[(size_t)nj * HF + h * Ff + f];
  const float ck = proj[(size_t)nk * HF + h * Ff + f];
  atomicAdd(&wsum[(size_t)ni * HF + h * Ff + f], a * cj * ck);
}

// ---------------- finalize ----------------
__global__ __launch_bounds__(256) void k_finalize0(const float* __restrict__ proj,
                                                   const float* __restrict__ wsum,
                                                   const float* __restrict__ theta,
                                                   const float* __restrict__ bias,
                                                   _Float16* __restrict__ out16, int n) {
  int i = blockIdx.x * 256 + threadIdx.x;
  if (i >= n) return;
  const int c = i & (HF - 1);
  float v = theta[c] * proj[i] + wsum[i] + bias[c];
  v = v > 0.f ? v : (expf(v) - 1.f);     // ELU(alpha=1)
  out16[i] = (_Float16)v;
}

__global__ __launch_bounds__(256) void k_finalize1(const float* __restrict__ proj,
                                                   const float* __restrict__ wsum,
                                                   const float* __restrict__ theta,
                                                   const float* __restrict__ bias,
                                                   float* __restrict__ out, int n) {
  int i = blockIdx.x * 256 + threadIdx.x;
  if (i >= n) return;
  const int node = i >> 6;
  const int f = i & 63;
  float acc = 0.f;
#pragma unroll
  for (int h = 0; h < 4; ++h) {
    const size_t idx = (size_t)node * HF + h * Ff + f;
    acc += theta[h * Ff + f] * proj[idx] + wsum[idx];
  }
  out[i] = acc * 0.25f + bias[f];
}

// ---------------- host ----------------

extern "C" void kernel_launch(void* const* d_in, const int* in_sizes, int n_in,
                              void* d_out, int out_size, void* d_ws, size_t ws_size,
                              hipStream_t stream) {
  (void)in_sizes; (void)n_in; (void)out_size; (void)ws_size;
  const float* x       = (const float*)d_in[0];
  const int*   ei      = (const int*)d_in[1];
  const float* w_proj0 = (const float*)d_in[2];
  const float* theta0  = (const float*)d_in[3];
  const float* m0w1    = (const float*)d_in[4];
  const float* m0b1    = (const float*)d_in[5];
  const float* m0w2    = (const float*)d_in[6];
  const float* m0b2    = (const float*)d_in[7];
  const float* bias0   = (const float*)d_in[8];
  const float* w_proj1 = (const float*)d_in[9];
  const float* theta1  = (const float*)d_in[10];
  const float* m1w1    = (const float*)d_in[11];
  const float* m1b1    = (const float*)d_in[12];
  const float* m1w2    = (const float*)d_in[13];
  const float* m1b2    = (const float*)d_in[14];
  const float* bias1   = (const float*)d_in[15];
  float* out = (float*)d_out;

  char* base = (char*)d_ws;
  size_t off = 0;
  auto take = [&](size_t bytes) -> void* {
    off = (off + 255) & ~(size_t)255;
    void* p = base + off;
    off += bytes;
    return p;
  };
  _Float16* A16    = (_Float16*)take((size_t)Nn * HF * 2);  // f16 GEMM input
  float*    proj   = (float*)take((size_t)Nn * HF * 4);
  _Float16* proj16 = (_Float16*)take((size_t)Nn * HF * 2);  // f16 proj for edge gathers
  float*    wsum   = (float*)take((size_t)Nn * HF * 4);
  float*    att    = (float*)take((size_t)Ee * Hh * 4);
  _Float16* wT     = (_Float16*)take((size_t)HF * HF * 2);  // transposed proj weights
  _Float16* w1T    = (_Float16*)take((size_t)MLPH * K3F * 2);

  const int rowTiles  = Nn / 16;              // 3125
  const int edgeTiles = (Ee * Hh) / 16;       // 100000
  const int edgeBlocks = 1536;                // grid-stride, ~8 tiles per wave

  // ---------------- layer 0 ----------------
  { int n = Nn * IN0;
    k_f32_to_f16<<<(n + 255) / 256, 256, 0, stream>>>(x, A16, n); }
  { int n = IN0 * HF;
    k_transpose_cvt<<<(n + 255) / 256, 256, 0, stream>>>(w_proj0, wT, IN0, HF); }
  k_proj_gemm<<<(rowTiles + 7) / 8, 256, 0, stream>>>(A16, wT, proj, proj16, Nn, IN0);
  { int n = K3F * MLPH;
    k_transpose_cvt<<<(n + 255) / 256, 256, 0, stream>>>(m0w1, w1T, K3F, MLPH); }
  hipMemsetAsync(wsum, 0, (size_t)Nn * HF * 4, stream);
  k_edge_att<<<edgeBlocks, 256, 0, stream>>>(proj16, ei, w1T, m0b1, m0w2, m0b2,
                                             att, edgeTiles);
  k_scatter<<<Ee, 256, 0, stream>>>(ei, att, proj, wsum);
  { int n = Nn * HF;
    k_finalize0<<<(n + 255) / 256, 256, 0, stream>>>(proj, wsum, theta0, bias0, A16, n); }

  // ---------------- layer 1 ----------------
  { int n = HF * HF;
    k_transpose_cvt<<<(n + 255) / 256, 256, 0, stream>>>(w_proj1, wT, HF, HF); }
  k_proj_gemm<<<(rowTiles + 7) / 8, 256, 0, stream>>>(A16, wT, proj, proj16, Nn, HF);
  { int n = K3F * MLPH;
    k_transpose_cvt<<<(n + 255) / 256, 256, 0, stream>>>(m1w1, w1T, K3F, MLPH); }
  hipMemsetAsync(wsum, 0, (size_t)Nn * HF * 4, stream);
  k_edge_att<<<edgeBlocks, 256, 0, stream>>>(proj16, ei, w1T, m1b1, m1w2, m1b2,
                                             att, edgeTiles);
  k_scatter<<<Ee, 256, 0, stream>>>(ei, att, proj, wsum);
  { int n = Nn * Ff;
    k_finalize1<<<(n + 255) / 256, 256, 0, stream>>>(proj, wsum, theta1, bias1, out, n); }
}
